// GCNwithJK_1623497638186
// MI455X (gfx1250) — matrix operand order repaired and checked by reading, not compiled, not run on hardware
//
#include <hip/hip_runtime.h>
#include <hip/hip_bf16.h>

// ---------------------------------------------------------------------------
// GCN-with-JumpingKnowledge forward for MI455X (gfx1250, wave32, WMMA).
//
// Pipeline:
//   h   = x @ W1                      (WMMA bf16 in / f32 acc)
//   agg1= scatter_add(h[src] -> dst)  (global_atomic_add_f32, L2-resident)
//   h1  = relu(BN(agg1 + b1))         (two-pass: stats -> finalize -> apply)
//   t   = h1 @ W2                     (WMMA)
//   agg2= scatter_add(t[src] -> dst)
//   out = max(h1, agg2 + b2) @ Wfc + bfc   (fused into GEMM3 A-load/store)
// ---------------------------------------------------------------------------

typedef __attribute__((ext_vector_type(16))) __bf16 v16bf;
typedef __attribute__((ext_vector_type(8)))  float  v8f;

#define N_COLS 64   // H == OUT == 64 everywhere after layer 1

// ---------------------------------------------------------------------------
// Zero a region of floats (count must be a multiple of 4).
// ---------------------------------------------------------------------------
__global__ void zero_f32_kernel(float4* __restrict__ p, long n4) {
    long i = (long)blockIdx.x * blockDim.x + threadIdx.x;
    long stride = (long)gridDim.x * blockDim.x;
    float4 z = make_float4(0.f, 0.f, 0.f, 0.f);
    for (; i < n4; i += stride) p[i] = z;
}

// ---------------------------------------------------------------------------
// WMMA GEMM: out[rows x 64] = A[rows x K] @ W[K x 64]
// One wave computes a 16-row x 64-col stripe. K in {128, 64}, multiple of 32.
// A/B are f32 in memory, converted to bf16 per-lane into the ISA register
// striping for v_wmma_f32_16x16x32_bf16.
// ---------------------------------------------------------------------------
template<int K>
__global__ void gemm_bf16_kernel(const float* __restrict__ A,
                                 const float* __restrict__ W,
                                 float* __restrict__ out,
                                 int nrows) {
    const int lane = threadIdx.x & 31;
    const int waveId = (int)((blockIdx.x * blockDim.x + threadIdx.x) >> 5);
    const int row0 = waveId * 16;
    if (row0 >= nrows) return;
    const int half = lane >> 4;     // 0: lanes 0-15, 1: lanes 16-31
    const int l16  = lane & 15;
    const int arow = row0 + l16;

    v8f c[4] = {v8f{}, v8f{}, v8f{}, v8f{}};

    for (int kb = 0; kb < K; kb += 32) {
        // A tile, 16x32 bf16: lane l16 = row; element e -> K offset per ISA layout.
        v16bf a;
#pragma unroll
        for (int e = 0; e < 16; ++e) {
            int k = kb + ((e < 8) ? 0 : 16) + half * 8 + (e & 7);
            a[e] = (__bf16)A[(long)arow * K + k];
        }
        // 4 B tiles (32x16 bf16 each): lane = column, element e -> K.
#pragma unroll
        for (int t = 0; t < 4; ++t) {
            v16bf b;
            int n = t * 16 + l16;
#pragma unroll
            for (int e = 0; e < 16; ++e) {
                int k = kb + ((e < 8) ? 0 : 16) + half * 8 + (e & 7);
                b[e] = (__bf16)W[k * N_COLS + n];
            }
            c[t] = __builtin_amdgcn_wmma_f32_16x16x32_bf16(
                false, a, false, b, (short)0, c[t], false, false);
        }
    }

    // D layout: lane l -> col (l&15); VGPR r -> row r + 8*half.
#pragma unroll
    for (int r = 0; r < 8; ++r) {
        int m = half * 8 + r;
        float* orow = out + (long)(row0 + m) * N_COLS;
        orow[ 0 + l16] = c[0][r];
        orow[16 + l16] = c[1][r];
        orow[32 + l16] = c[2][r];
        orow[48 + l16] = c[3][r];
    }
}

// ---------------------------------------------------------------------------
// GEMM3 fused: out = max(h1, agg2 + b2) @ Wfc + bfc   (K = 64)
// ---------------------------------------------------------------------------
__global__ void gemm_jk_fused_kernel(const float* __restrict__ h1,
                                     const float* __restrict__ agg2,
                                     const float* __restrict__ b2,
                                     const float* __restrict__ Wfc,
                                     const float* __restrict__ bfc,
                                     float* __restrict__ out,
                                     int nrows) {
    const int K = 64;
    const int lane = threadIdx.x & 31;
    const int waveId = (int)((blockIdx.x * blockDim.x + threadIdx.x) >> 5);
    const int row0 = waveId * 16;
    if (row0 >= nrows) return;
    const int half = lane >> 4;
    const int l16  = lane & 15;
    const int arow = row0 + l16;

    v8f c[4] = {v8f{}, v8f{}, v8f{}, v8f{}};

    for (int kb = 0; kb < K; kb += 32) {
        v16bf a;
#pragma unroll
        for (int e = 0; e < 16; ++e) {
            int k = kb + ((e < 8) ? 0 : 16) + half * 8 + (e & 7);
            long idx = (long)arow * K + k;
            float v1 = h1[idx];
            float v2 = agg2[idx] + b2[k];
            a[e] = (__bf16)fmaxf(v1, v2);
        }
#pragma unroll
        for (int t = 0; t < 4; ++t) {
            v16bf b;
            int n = t * 16 + l16;
#pragma unroll
            for (int e = 0; e < 16; ++e) {
                int k = kb + ((e < 8) ? 0 : 16) + half * 8 + (e & 7);
                b[e] = (__bf16)Wfc[k * N_COLS + n];
            }
            c[t] = __builtin_amdgcn_wmma_f32_16x16x32_bf16(
                false, a, false, b, (short)0, c[t], false, false);
        }
    }

#pragma unroll
    for (int r = 0; r < 8; ++r) {
        int m = half * 8 + r;
        float* orow = out + (long)(row0 + m) * N_COLS;
        orow[ 0 + l16] = c[0][r] + bfc[ 0 + l16];
        orow[16 + l16] = c[1][r] + bfc[16 + l16];
        orow[32 + l16] = c[2][r] + bfc[32 + l16];
        orow[48 + l16] = c[3][r] + bfc[48 + l16];
    }
}

// ---------------------------------------------------------------------------
// Edge scatter-add: one wave per edge (grid-stride). Lane j handles feature
// columns j and j+32: 2x128B coalesced gather + 2x128B coalesced f32 atomics.
// agg (25.6 MB) is L2-resident, so the RMW resolves in L2.
// ---------------------------------------------------------------------------
__global__ void scatter_add_kernel(const float* __restrict__ h,
                                   const int* __restrict__ src,
                                   const int* __restrict__ dst,
                                   float* __restrict__ agg,
                                   int nedges) {
    const int lane = threadIdx.x & 31;
    int wave   = (int)((blockIdx.x * blockDim.x + threadIdx.x) >> 5);
    int nwaves = (int)((gridDim.x * blockDim.x) >> 5);
    for (int e = wave; e < nedges; e += nwaves) {
        int s = src[e];
        int d = dst[e];
        float v0 = h[(long)s * N_COLS + lane];
        float v1 = h[(long)s * N_COLS + 32 + lane];
        unsafeAtomicAdd(&agg[(long)d * N_COLS + lane], v0);
        unsafeAtomicAdd(&agg[(long)d * N_COLS + 32 + lane], v1);
    }
}

// ---------------------------------------------------------------------------
// BatchNorm column statistics: sum and sum-of-squares of (agg + b1) per column.
// 256 threads = 4 rows per iteration; LDS tree-combine; atomic into 64-wide
// global accumulators.
// ---------------------------------------------------------------------------
__global__ void bn_stats_kernel(const float* __restrict__ agg,
                                const float* __restrict__ b1,
                                float* __restrict__ sums,
                                float* __restrict__ sumsq,
                                int nrows) {
    const int col  = threadIdx.x & 63;
    const int rsub = threadIdx.x >> 6;      // 0..3
    const float bias = b1[col];
    float s = 0.f, sq = 0.f;
    for (int r = blockIdx.x * 4 + rsub; r < nrows; r += gridDim.x * 4) {
        float v = agg[(long)r * N_COLS + col] + bias;
        s  += v;
        sq += v * v;
    }
    __shared__ float ls[256];
    __shared__ float lq[256];
    ls[threadIdx.x] = s;
    lq[threadIdx.x] = sq;
    __syncthreads();
    if (threadIdx.x < 64) {
        s  = ls[threadIdx.x] + ls[threadIdx.x + 64] + ls[threadIdx.x + 128] + ls[threadIdx.x + 192];
        sq = lq[threadIdx.x] + lq[threadIdx.x + 64] + lq[threadIdx.x + 128] + lq[threadIdx.x + 192];
        unsafeAtomicAdd(&sums[threadIdx.x], s);
        unsafeAtomicAdd(&sumsq[threadIdx.x], sq);
    }
}

__global__ void bn_finalize_kernel(const float* __restrict__ sums,
                                   const float* __restrict__ sumsq,
                                   const float* __restrict__ gamma,
                                   const float* __restrict__ beta,
                                   float* __restrict__ scale,
                                   float* __restrict__ shift,
                                   float invN) {
    int c = threadIdx.x;              // 64 threads
    float mean = sums[c] * invN;
    float var  = sumsq[c] * invN - mean * mean;   // biased variance
    float sc   = gamma[c] * rsqrtf(var + 1e-5f);
    scale[c] = sc;
    shift[c] = beta[c] - mean * sc;
}

__global__ void bn_apply_relu_kernel(const float* __restrict__ agg,
                                     const float* __restrict__ b1,
                                     const float* __restrict__ scale,
                                     const float* __restrict__ shift,
                                     float* __restrict__ h1,
                                     long total) {
    long i = (long)blockIdx.x * blockDim.x + threadIdx.x;
    long stride = (long)gridDim.x * blockDim.x;
    for (; i < total; i += stride) {
        int c = (int)(i & 63);
        float v = (agg[i] + b1[c]) * scale[c] + shift[c];
        h1[i] = v > 0.f ? v : 0.f;
    }
}

// ---------------------------------------------------------------------------
// Host-side orchestration.
// Input order: x, edge_index, W1, b1, gamma, beta, W2, b2, Wfc, bfc
// ---------------------------------------------------------------------------
extern "C" void kernel_launch(void* const* d_in, const int* in_sizes, int n_in,
                              void* d_out, int out_size, void* d_ws, size_t ws_size,
                              hipStream_t stream) {
    const float* x     = (const float*)d_in[0];
    const int*   eidx  = (const int*)  d_in[1];
    const float* W1    = (const float*)d_in[2];
    const float* b1    = (const float*)d_in[3];
    const float* gamma = (const float*)d_in[4];
    const float* beta  = (const float*)d_in[5];
    const float* W2    = (const float*)d_in[6];
    const float* b2    = (const float*)d_in[7];
    const float* Wfc   = (const float*)d_in[8];
    const float* bfc   = (const float*)d_in[9];
    float* out = (float*)d_out;

    const int IN = 128;
    const int N  = in_sizes[0] / IN;       // 100000
    const int E  = in_sizes[1] / 2;        // 1600000
    const int* src = eidx;
    const int* dst = eidx + E;

    const long N64 = (long)N * N_COLS;

    // Workspace layout (floats): agg1 | agg2 | stats(256) | h | h1
    float* agg1  = (float*)d_ws;
    float* agg2  = agg1 + N64;
    float* stats = agg2 + N64;             // sums[64] sumsq[64] scale[64] shift[64]
    float* sums  = stats;
    float* sumsq = stats + 64;
    float* scale = stats + 128;
    float* shift = stats + 192;
    float* h     = stats + 256;            // GEMM1 output, reused for GEMM2 output
    float* h1    = h + N64;

    // 1) Zero accumulators + stats (contiguous region: 2*N64 + 256 floats).
    {
        long n4 = (2 * N64 + 256) / 4;
        zero_f32_kernel<<<2048, 256, 0, stream>>>((float4*)agg1, n4);
    }

    const int gemmBlocks = (N / 16 + 7) / 8;   // 8 waves (16 rows each) per block

    // 2) h = x @ W1
    gemm_bf16_kernel<128><<<gemmBlocks, 256, 0, stream>>>(x, W1, h, N);

    // 3) agg1 = scatter_add(h[src] -> dst)
    scatter_add_kernel<<<8192, 256, 0, stream>>>(h, src, dst, agg1, E);

    // 4) BN stats over (agg1 + b1)
    bn_stats_kernel<<<1024, 256, 0, stream>>>(agg1, b1, sums, sumsq, N);
    bn_finalize_kernel<<<1, 64, 0, stream>>>(sums, sumsq, gamma, beta, scale, shift,
                                             1.0f / (float)N);

    // 5) h1 = relu(BN(agg1 + b1))
    bn_apply_relu_kernel<<<4096, 256, 0, stream>>>(agg1, b1, scale, shift, h1, N64);

    // 6) t = h1 @ W2   (reuses h buffer)
    gemm_bf16_kernel<64><<<gemmBlocks, 256, 0, stream>>>(h1, W2, h, N);

    // 7) agg2 = scatter_add(t[src] -> dst)
    scatter_add_kernel<<<8192, 256, 0, stream>>>(h, src, dst, agg2, E);

    // 8) out = max(h1, agg2 + b2) @ Wfc + bfc
    gemm_jk_fused_kernel<<<gemmBlocks, 256, 0, stream>>>(h1, agg2, b2, Wfc, bfc, out, N);
}
